// NeuralNetwork_86182813762422
// MI455X (gfx1250) — compile-verified, matrix operand order checked by reading, not run on hardware
//
#include <hip/hip_runtime.h>
#include <stdint.h>

#define B_  2
#define S_  4096
#define D_  768
#define H_  12
#define DH_ 64
#define FF_ 3072
#define L_  2
#define W_  256
#define GH_ 512

#define USE_ASYNC 1

typedef unsigned short u16;
typedef unsigned int   u32;
typedef __bf16 v16bf __attribute__((ext_vector_type(16)));
typedef float  v8f   __attribute__((ext_vector_type(8)));

struct U4x2 { uint4 a; uint4 b; };
union  V8   { v8f v; float f[8]; };

// ---------- helpers ----------
__device__ __forceinline__ u16 f2bf(float f) {
  u32 u = __builtin_bit_cast(u32, f);
  u32 r = u + 0x7FFFu + ((u >> 16) & 1u);
  return (u16)(r >> 16);
}
__device__ __forceinline__ float bf2f(u16 h) {
  u32 u = ((u32)h) << 16;
  return __builtin_bit_cast(float, u);
}
__device__ __forceinline__ v8f wmma_bf(v16bf a, v16bf b, v8f c) {
  return __builtin_amdgcn_wmma_f32_16x16x32_bf16(false, a, false, b, (short)0, c, false, false);
}
// Load an A/B-layout fragment: lane l -> row (l&15), K runs [8*hs..8*hs+7] and [16+8*hs..23+8*hs]
__device__ __forceinline__ v16bf frag_ld(const u16* p, int stride) {
  int l = threadIdx.x & 31;
  const u16* q = p + (size_t)(l & 15) * stride + (l >> 4) * 8;
  U4x2 t;
  t.a = *(const uint4*)(q);
  t.b = *(const uint4*)(q + 16);
  return __builtin_bit_cast(v16bf, t);
}
__device__ __forceinline__ float geluf(float x) {
  float x3 = x * x * x;
  return 0.5f * x * (1.f + tanhf(0.7978845608028654f * (x + 0.044715f * x3)));
}
__device__ __forceinline__ float sigmf(float x) { return 1.f / (1.f + __expf(-x)); }

#if USE_ASYNC
typedef __attribute__((address_space(3))) u16 lds_u16_t;
__device__ __forceinline__ u32 lds_off(const u16* p) {
  return (u32)(uintptr_t)(lds_u16_t*)(p);
}
#endif

// ---------- weight conversion ----------
__global__ void k_cvt(const float* __restrict__ in, u16* __restrict__ out, size_t n) {
  size_t i = (size_t)blockIdx.x * 256 + threadIdx.x;
  if (i < n) out[i] = f2bf(in[i]);
}
// in: R x C row-major (fp32)  ->  out: C x R row-major (bf16)   (i.e. out = in^T)
__global__ void k_cvt_t(const float* __restrict__ in, u16* __restrict__ out, int R, int C) {
  size_t i = (size_t)blockIdx.x * 256 + threadIdx.x;
  if (i < (size_t)R * C) {
    int r = (int)(i / C), c = (int)(i % C);
    out[(size_t)c * R + r] = f2bf(in[i]);
  }
}

// ---------- embedding ----------
__global__ void k_embed(const float* __restrict__ ew, const float* __restrict__ ep,
                        const float* __restrict__ et, const int* __restrict__ ids,
                        const int* __restrict__ tts, float* __restrict__ x) {
  int row = blockIdx.x;            // b*S + s
  int s = row % S_;
  int id = ids[row], tt = tts[row];
  for (int d = threadIdx.x; d < D_; d += blockDim.x)
    x[(size_t)row * D_ + d] = ew[(size_t)id * D_ + d] + ep[(size_t)s * D_ + d] + et[(size_t)tt * D_ + d];
}

// ---------- layernorm (row = block), writes fp32 + bf16 ----------
__global__ void __launch_bounds__(256) k_ln(const float* __restrict__ in,
                                            const float* __restrict__ g,
                                            const float* __restrict__ bta,
                                            float* __restrict__ outf,
                                            u16* __restrict__ outb) {
  int row = blockIdx.x, tid = threadIdx.x;
  const float* x = in + (size_t)row * D_;
  float v0 = x[tid], v1 = x[tid + 256], v2 = x[tid + 512];
  __shared__ float red[8];
  float s = v0 + v1 + v2;
  for (int m = 1; m < 32; m <<= 1) s += __shfl_xor(s, m, 32);
  if ((tid & 31) == 0) red[tid >> 5] = s;
  __syncthreads();
  float tot = 0.f;
  for (int i = 0; i < 8; ++i) tot += red[i];
  float mean = tot * (1.f / 768.f);
  float d0 = v0 - mean, d1 = v1 - mean, d2 = v2 - mean;
  float q = d0 * d0 + d1 * d1 + d2 * d2;
  for (int m = 1; m < 32; m <<= 1) q += __shfl_xor(q, m, 32);
  __syncthreads();
  if ((tid & 31) == 0) red[tid >> 5] = q;
  __syncthreads();
  float qt = 0.f;
  for (int i = 0; i < 8; ++i) qt += red[i];
  float rstd = rsqrtf(qt * (1.f / 768.f) + 1e-12f);
  float o0 = d0 * rstd * g[tid]       + bta[tid];
  float o1 = d1 * rstd * g[tid + 256] + bta[tid + 256];
  float o2 = d2 * rstd * g[tid + 512] + bta[tid + 512];
  size_t base = (size_t)row * D_;
  outf[base + tid] = o0; outf[base + tid + 256] = o1; outf[base + tid + 512] = o2;
  outb[base + tid] = f2bf(o0); outb[base + tid + 256] = f2bf(o1); outb[base + tid + 512] = f2bf(o2);
}

// ---------- generic bf16 WMMA GEMM (templated epilogue, no spills) ----------
// C = act(A[MxK] * B[KxN] + bias (+ resid)). Block tile 128x128, 8 waves, wave tile 32x64.
template <int ACT, bool RES, bool OUTF, bool OUTB>
__global__ void __launch_bounds__(256) k_gemm_t(const u16* __restrict__ A,
                                                const u16* __restrict__ Bw,
                                                const float* __restrict__ bias,
                                                const float* __restrict__ resid,
                                                float* __restrict__ Cf,
                                                u16* __restrict__ Cb,
                                                int M, int N, int K) {
  __shared__ __align__(16) u16 As[128 * 32];
  __shared__ __align__(16) u16 Bs[128 * 32];   // transposed: Bs[n][k]
  int tid = threadIdx.x, wave = tid >> 5, lane = tid & 31;
  int bm = blockIdx.y * 128, bn = blockIdx.x * 128;
  int wm = (wave >> 1) * 32, wn = (wave & 1) * 64;
  V8 acc[2][4];
#pragma unroll
  for (int i = 0; i < 2; ++i)
#pragma unroll
    for (int j = 0; j < 4; ++j)
#pragma unroll
      for (int r = 0; r < 8; ++r) acc[i][j].f[r] = 0.f;

  // per-thread fixed tile roles
  int arow = tid >> 1, aoff = (tid & 1) * 16;        // A: 128 rows x 32, 16 elems/thread
  const u16* gA = A + (size_t)(bm + arow) * K + aoff;
  int bk = tid >> 3, bnOff = (tid & 7) * 16;         // B: 32 rows x 128, 16 elems/thread
  const u16* gB = Bw + (size_t)bk * N + bn + bnOff;
#if USE_ASYNC
  u32 aLds = lds_off(As + arow * 32 + aoff);
#else
  uint4* aSp = (uint4*)(As + arow * 32 + aoff);
#endif

  for (int k0 = 0; k0 < K; k0 += 32) {
#if USE_ASYNC
    { // A tile via async DMA to LDS (16B per lane, twice)
      unsigned long long ga = (unsigned long long)(uintptr_t)(gA + k0);
      asm volatile("global_load_async_to_lds_b128 %0, %1, off"
                   :: "v"(aLds), "v"(ga) : "memory");
      asm volatile("global_load_async_to_lds_b128 %0, %1, off offset:16"
                   :: "v"(aLds), "v"(ga) : "memory");
    }
#else
    {
      const uint4* g = (const uint4*)(gA + k0);
      uint4 a0 = g[0], a1 = g[1];
      aSp[0] = a0; aSp[1] = a1;
    }
#endif
    { // B tile 32x128 -> Bs transposed
      const u16* gb = gB + (size_t)k0 * N;
      __builtin_prefetch(gb + (size_t)32 * N, 0, 1);
      uint4 b0 = ((const uint4*)gb)[0], b1 = ((const uint4*)gb)[1];
      const u16* bv0 = (const u16*)&b0;
      const u16* bv1 = (const u16*)&b1;
#pragma unroll
      for (int j = 0; j < 8; ++j) Bs[(bnOff + j) * 32 + bk] = bv0[j];
#pragma unroll
      for (int j = 0; j < 8; ++j) Bs[(bnOff + 8 + j) * 32 + bk] = bv1[j];
    }
#if USE_ASYNC
    asm volatile("s_wait_asynccnt 0" ::: "memory");
#endif
    __syncthreads();
    v16bf a0 = frag_ld(As + (size_t)wm * 32, 32);
    v16bf a1 = frag_ld(As + (size_t)(wm + 16) * 32, 32);
#pragma unroll
    for (int j = 0; j < 4; ++j) {
      v16bf bf = frag_ld(Bs + (size_t)(wn + j * 16) * 32, 32);
      acc[0][j].v = wmma_bf(a0, bf, acc[0][j].v);
      acc[1][j].v = wmma_bf(a1, bf, acc[1][j].v);
    }
    __syncthreads();
  }

  int hs = lane >> 4, ln = lane & 15;
#pragma unroll
  for (int i = 0; i < 2; ++i)
#pragma unroll
    for (int j = 0; j < 4; ++j) {
      int col = bn + wn + j * 16 + ln;
      float bb = bias[col];
#pragma unroll
      for (int r = 0; r < 8; ++r) {
        int row = bm + wm + i * 16 + r + 8 * hs;
        float v = acc[i][j].f[r] + bb;
        if (RES) v += resid[(size_t)row * N + col];
        if (ACT == 1) v = geluf(v);
        if (OUTF) Cf[(size_t)row * N + col] = v;
        if (OUTB) Cb[(size_t)row * N + col] = f2bf(v);
      }
    }
}

// ---------- sliding-window flash attention ----------
// grid (nc=16, H, B), block 256 (8 waves). Wave w handles q rows [w*32, w*32+32).
__global__ void __launch_bounds__(256) k_attn(const u16* __restrict__ qb,
                                              const u16* __restrict__ kb,
                                              const u16* __restrict__ vb,
                                              const int* __restrict__ amask,
                                              u16* __restrict__ ob) {
  __shared__ __align__(16) u16 Kl[64 * 64];     // [key][dh]
  __shared__ __align__(16) u16 Vt[64 * 64];     // [dh][key]
  __shared__ __align__(16) u16 Pl[256 * 64];    // probs per block, [qrow][key]
  __shared__ float ml[64];
  int c = blockIdx.x, h = blockIdx.y, b = blockIdx.z;
  int tid = threadIdx.x, wave = tid >> 5, lane = tid & 31;
  int hs = lane >> 4, ln = lane & 15;
  int q0 = c * W_ + wave * 32;

  v16bf aq[2][2];
  for (int i = 0; i < 2; ++i)
    for (int kk = 0; kk < 2; ++kk)
      aq[i][kk] = frag_ld(qb + ((size_t)(b * S_ + q0 + i * 16)) * D_ + h * DH_ + kk * 32, D_);

  V8 o[2][4];
  float rm[2][8], rs[2][8];
  for (int i = 0; i < 2; ++i)
    for (int r = 0; r < 8; ++r) {
      rm[i][r] = -1e30f; rs[i][r] = 0.f;
      for (int cg = 0; cg < 4; ++cg) o[i][cg].f[r] = 0.f;
    }

  for (int kt = 0; kt < 12; ++kt) {
    int kbase = c * W_ - W_ + kt * 64;
    { // load K tile + V^T tile
      int row = tid >> 2, chunk = tid & 3;
      int kg = kbase + row;
      uint4 d0 = {}, d1 = {}, e0 = {}, e1 = {};
      if (kg >= 0 && kg < S_) {
        const uint4* gk = (const uint4*)(kb + ((size_t)(b * S_ + kg)) * D_ + h * DH_ + chunk * 16);
        d0 = gk[0]; d1 = gk[1];
        const uint4* gv = (const uint4*)(vb + ((size_t)(b * S_ + kg)) * D_ + h * DH_ + chunk * 16);
        e0 = gv[0]; e1 = gv[1];
      }
      *(uint4*)(Kl + row * 64 + chunk * 16)     = d0;
      *(uint4*)(Kl + row * 64 + chunk * 16 + 8) = d1;
      const u16* ev = (const u16*)&e0;
#pragma unroll
      for (int j = 0; j < 8; ++j) Vt[(chunk * 16 + j) * 64 + row] = ev[j];
      ev = (const u16*)&e1;
#pragma unroll
      for (int j = 0; j < 8; ++j) Vt[(chunk * 16 + 8 + j) * 64 + row] = ev[j];
    }
    if (tid < 64) {
      int kg = kbase + tid;
      ml[tid] = (kg >= 0 && kg < S_ && amask[b * S_ + kg] > 0) ? 0.f : -1e9f;
    }
    __syncthreads();

    // scores = Q * K^T (scaled), masked
    V8 sc[2][4];
    for (int j = 0; j < 4; ++j) {
      v16bf bk0 = frag_ld(Kl + (size_t)(j * 16) * 64, 64);
      v16bf bk1 = frag_ld(Kl + (size_t)(j * 16) * 64 + 32, 64);
      for (int i = 0; i < 2; ++i) {
        V8 z; for (int r = 0; r < 8; ++r) z.f[r] = 0.f;
        z.v = wmma_bf(aq[i][0], bk0, z.v);
        z.v = wmma_bf(aq[i][1], bk1, z.v);
        sc[i][j] = z;
      }
    }
    for (int i = 0; i < 2; ++i) {
      float tmax[8];
#pragma unroll
      for (int r = 0; r < 8; ++r) tmax[r] = -1e30f;
      for (int j = 0; j < 4; ++j) {
        int kg = kbase + j * 16 + ln;
        float mj = ml[j * 16 + ln];
#pragma unroll
        for (int r = 0; r < 8; ++r) {
          int qg = q0 + i * 16 + r + 8 * hs;
          float v = sc[i][j].f[r] * 0.125f + mj;
          int dd = kg - qg;
          if (dd < -W_ || dd > W_) v += -1e9f;
          sc[i][j].f[r] = v;
          tmax[r] = fmaxf(tmax[r], v);
        }
      }
#pragma unroll
      for (int r = 0; r < 8; ++r) {
        float t = tmax[r];
        for (int m = 1; m < 16; m <<= 1) t = fmaxf(t, __shfl_xor(t, m, 32));
        float mnew = fmaxf(rm[i][r], t);
        float alpha = __expf(rm[i][r] - mnew);
        rm[i][r] = mnew;
        rs[i][r] *= alpha;
        for (int cg = 0; cg < 4; ++cg) o[i][cg].f[r] *= alpha;
        float ps = 0.f;
        for (int j = 0; j < 4; ++j) {
          float p = __expf(sc[i][j].f[r] - mnew);
          ps += p;
          Pl[(size_t)(wave * 32 + i * 16 + r + 8 * hs) * 64 + j * 16 + ln] = f2bf(p);
        }
        for (int m = 1; m < 16; m <<= 1) ps += __shfl_xor(ps, m, 32);
        rs[i][r] += ps;
      }
    }
    // O += P * V
    for (int i = 0; i < 2; ++i)
      for (int kk = 0; kk < 2; ++kk) {
        v16bf ap = frag_ld(Pl + (size_t)(wave * 32 + i * 16) * 64 + kk * 32, 64);
        for (int cg = 0; cg < 4; ++cg) {
          v16bf bv = frag_ld(Vt + (size_t)(cg * 16) * 64 + kk * 32, 64);
          o[i][cg].v = wmma_bf(ap, bv, o[i][cg].v);
        }
      }
    __syncthreads();
  }
  for (int i = 0; i < 2; ++i)
    for (int cg = 0; cg < 4; ++cg)
#pragma unroll
      for (int r = 0; r < 8; ++r) {
        int qg = q0 + i * 16 + r + 8 * hs;
        float v = o[i][cg].f[r] / rs[i][r];
        ob[((size_t)(b * S_ + qg)) * D_ + h * DH_ + cg * 16 + ln] = f2bf(v);
      }
}

// ---------- GRU forward recurrence (persistent single workgroup) ----------
__global__ void __launch_bounds__(1024) k_gru_f(const float* __restrict__ gi,
                                                const u16* __restrict__ whh,
                                                const float* __restrict__ bhh,
                                                float* __restrict__ hout) {
  __shared__ float h[2 * GH_];
  int tid = threadIdx.x;
  h[tid] = 0.f;
  __syncthreads();
  int b = tid >> 9, n = tid & 511;
  const float* hb = h + b * GH_;
  for (int t = 0; t < S_; ++t) {
    float hr = bhh[n], hz = bhh[GH_ + n], hn = bhh[2 * GH_ + n];
    const u16* wr = whh + (size_t)n * GH_;
    const u16* wz = whh + (size_t)(GH_ + n) * GH_;
    const u16* wn = whh + (size_t)(2 * GH_ + n) * GH_;
    for (int k = 0; k < GH_; ++k) {
      float hk = hb[k];
      hr += hk * bf2f(wr[k]);
      hz += hk * bf2f(wz[k]);
      hn += hk * bf2f(wn[k]);
    }
    const float* gir = gi + ((size_t)b * S_ + t) * (3 * GH_);
    float r  = sigmf(gir[n] + hr);
    float z  = sigmf(gir[GH_ + n] + hz);
    float nn = tanhf(gir[2 * GH_ + n] + r * hn);
    float hnew = (1.f - z) * nn + z * hb[n];
    __syncthreads();
    h[tid] = hnew;
    __syncthreads();
  }
  hout[tid] = h[tid];
}

// ---------- backward GRU: only t = S-1 matters (h0 = 0) ----------
__global__ void __launch_bounds__(256) k_gru_b1(const float* __restrict__ x,
                                                const float* __restrict__ wih,
                                                const float* __restrict__ bih,
                                                const float* __restrict__ bhh,
                                                float* __restrict__ hout) {
  int idx = blockIdx.x * 256 + threadIdx.x;   // 0..1023
  int b = idx >> 9, n = idx & 511;
  const float* xr = x + ((size_t)b * S_ + (S_ - 1)) * D_;
  float ir = bih[n], iz = bih[GH_ + n], in_ = bih[2 * GH_ + n];
  const float* w0 = wih + (size_t)n * D_;
  const float* w1 = wih + (size_t)(GH_ + n) * D_;
  const float* w2 = wih + (size_t)(2 * GH_ + n) * D_;
  for (int k = 0; k < D_; ++k) {
    float xv = xr[k];
    ir += xv * w0[k]; iz += xv * w1[k]; in_ += xv * w2[k];
  }
  float r  = sigmf(ir + bhh[n]);
  float z  = sigmf(iz + bhh[GH_ + n]);
  float nn = tanhf(in_ + r * bhh[2 * GH_ + n]);
  hout[idx] = (1.f - z) * nn;
}

// ---------- head ----------
__global__ void __launch_bounds__(256) k_head(const float* __restrict__ hf,
                                              const float* __restrict__ hb,
                                              const float* __restrict__ W3,
                                              const float* __restrict__ b3,
                                              const float* __restrict__ W4,
                                              const float* __restrict__ b4,
                                              float* __restrict__ out) {
  __shared__ float h3[2 * 512];
  int tid = threadIdx.x;
  for (int j = tid; j < 1024; j += 256) {
    int b = j >> 9, jj = j & 511;
    float acc = b3[jj];
    for (int k = 0; k < 512; ++k) acc += hf[b * 512 + k] * W3[(size_t)k * 512 + jj];
    for (int k = 0; k < 512; ++k) acc += hb[b * 512 + k] * W3[(size_t)(512 + k) * 512 + jj];
    h3[j] = fmaxf(acc, 0.f);
  }
  __syncthreads();
  if (tid < 4) {
    int b = tid >> 1, oo = tid & 1;
    float acc = b4[oo];
    for (int j = 0; j < 512; ++j) acc += h3[b * 512 + j] * W4[j * 2 + oo];
    out[b * 2 + oo] = acc;
  }
}

// ---------- host ----------
extern "C" void kernel_launch(void* const* d_in, const int* in_sizes, int n_in,
                              void* d_out, int out_size, void* d_ws, size_t ws_size,
                              hipStream_t stream) {
  (void)in_sizes; (void)n_in; (void)out_size; (void)ws_size;
  const int M = B_ * S_;

  const float* emb_word = (const float*)d_in[0];
  const float* emb_pos  = (const float*)d_in[1];
  const float* emb_type = (const float*)d_in[2];
  const float* emb_ln_g = (const float*)d_in[3];
  const float* emb_ln_b = (const float*)d_in[4];
  auto LW = [&](int l, int j) { return (const float*)d_in[5 + 16 * l + j]; };
  const float* gf_Wih = (const float*)d_in[37];
  const float* gf_Whh = (const float*)d_in[38];
  const float* gf_bih = (const float*)d_in[39];
  const float* gf_bhh = (const float*)d_in[40];
  const float* gb_Wih = (const float*)d_in[41];
  const float* gb_bih = (const float*)d_in[43];
  const float* gb_bhh = (const float*)d_in[44];
  const float* W3 = (const float*)d_in[45];
  const float* b3 = (const float*)d_in[46];
  const float* W4 = (const float*)d_in[47];
  const float* b4 = (const float*)d_in[48];
  const int* ids = (const int*)d_in[49];
  const int* tts = (const int*)d_in[50];
  const int* am  = (const int*)d_in[51];
  float* out = (float*)d_out;

  char* wsb = (char*)d_ws;
  size_t off = 0;
  auto alloc = [&](size_t bytes) -> void* {
    void* p = (void*)(wsb + off);
    off += (bytes + 255) & ~(size_t)255;
    return p;
  };
  float* x_f  = (float*)alloc((size_t)M * D_ * 4);
  float* xt_f = (float*)alloc((size_t)M * D_ * 4);
  u16* xb  = (u16*)alloc((size_t)M * D_ * 2);
  u16* qb  = (u16*)alloc((size_t)M * D_ * 2);
  u16* kb  = (u16*)alloc((size_t)M * D_ * 2);
  u16* vb  = (u16*)alloc((size_t)M * D_ * 2);
  u16* ab  = (u16*)alloc((size_t)M * D_ * 2);
  u16* h1b = (u16*)alloc((size_t)M * FF_ * 2);
  float* gi_f = (float*)alloc((size_t)M * 3 * GH_ * 4);
  u16* wqb = (u16*)alloc((size_t)D_ * D_ * 2);
  u16* wkb = (u16*)alloc((size_t)D_ * D_ * 2);
  u16* wvb = (u16*)alloc((size_t)D_ * D_ * 2);
  u16* wob = (u16*)alloc((size_t)D_ * D_ * 2);
  u16* w1b = (u16*)alloc((size_t)D_ * FF_ * 2);
  u16* w2b = (u16*)alloc((size_t)FF_ * D_ * 2);
  u16* wihT = (u16*)alloc((size_t)D_ * 3 * GH_ * 2);
  u16* whhb = (u16*)alloc((size_t)3 * GH_ * GH_ * 2);
  float* h_f = (float*)alloc(2 * GH_ * 4);
  float* h_b = (float*)alloc(2 * GH_ * 4);

  auto cvt = [&](const float* in, u16* o, size_t n) {
    k_cvt<<<(unsigned)((n + 255) / 256), 256, 0, stream>>>(in, o, n);
  };
  // variants: bf16-out, resid+f32-out, gelu+bf16-out, f32-out
  auto gemm_b = [&](const u16* A, const u16* Bw, const float* bias, u16* Cb, int m, int n, int k) {
    dim3 g(n / 128, m / 128);
    k_gemm_t<0, false, false, true><<<g, 256, 0, stream>>>(A, Bw, bias, nullptr, nullptr, Cb, m, n, k);
  };
  auto gemm_rf = [&](const u16* A, const u16* Bw, const float* bias, const float* resid, float* Cf,
                     int m, int n, int k) {
    dim3 g(n / 128, m / 128);
    k_gemm_t<0, true, true, false><<<g, 256, 0, stream>>>(A, Bw, bias, resid, Cf, nullptr, m, n, k);
  };
  auto gemm_gb = [&](const u16* A, const u16* Bw, const float* bias, u16* Cb, int m, int n, int k) {
    dim3 g(n / 128, m / 128);
    k_gemm_t<1, false, false, true><<<g, 256, 0, stream>>>(A, Bw, bias, nullptr, nullptr, Cb, m, n, k);
  };
  auto gemm_ff = [&](const u16* A, const u16* Bw, const float* bias, float* Cf, int m, int n, int k) {
    dim3 g(n / 128, m / 128);
    k_gemm_t<0, false, true, false><<<g, 256, 0, stream>>>(A, Bw, bias, nullptr, Cf, nullptr, m, n, k);
  };

  // embeddings + LN
  k_embed<<<M, 256, 0, stream>>>(emb_word, emb_pos, emb_type, ids, tts, xt_f);
  k_ln<<<M, 256, 0, stream>>>(xt_f, emb_ln_g, emb_ln_b, x_f, xb);

  for (int l = 0; l < L_; ++l) {
    cvt(LW(l, 0), wqb, (size_t)D_ * D_);
    cvt(LW(l, 2), wkb, (size_t)D_ * D_);
    cvt(LW(l, 4), wvb, (size_t)D_ * D_);
    cvt(LW(l, 6), wob, (size_t)D_ * D_);
    gemm_b(xb, wqb, LW(l, 1), qb, M, D_, D_);
    gemm_b(xb, wkb, LW(l, 3), kb, M, D_, D_);
    gemm_b(xb, wvb, LW(l, 5), vb, M, D_, D_);
    dim3 ga(S_ / W_, H_, B_);
    k_attn<<<ga, 256, 0, stream>>>(qb, kb, vb, am, ab);
    gemm_rf(ab, wob, LW(l, 7), x_f, xt_f, M, D_, D_);
    k_ln<<<M, 256, 0, stream>>>(xt_f, LW(l, 8), LW(l, 9), x_f, xb);
    cvt(LW(l, 10), w1b, (size_t)D_ * FF_);
    gemm_gb(xb, w1b, LW(l, 11), h1b, M, FF_, D_);
    cvt(LW(l, 12), w2b, (size_t)FF_ * D_);
    gemm_rf(h1b, w2b, LW(l, 13), x_f, xt_f, M, D_, FF_);
    k_ln<<<M, 256, 0, stream>>>(xt_f, LW(l, 14), LW(l, 15), x_f, xb);
  }

  // GRU: forward input projection (WMMA), then persistent recurrence
  k_cvt_t<<<(unsigned)(((size_t)3 * GH_ * D_ + 255) / 256), 256, 0, stream>>>(gf_Wih, wihT, 3 * GH_, D_);
  gemm_ff(xb, wihT, gf_bih, gi_f, M, 3 * GH_, D_);
  cvt(gf_Whh, whhb, (size_t)3 * GH_ * GH_);
  k_gru_f<<<1, 1024, 0, stream>>>(gi_f, whhb, gf_bhh, h_f);
  // backward GRU: only last timestep reaches the head (h0 = 0, one step)
  k_gru_b1<<<4, 256, 0, stream>>>(x_f, gb_Wih, gb_bih, gb_bhh, h_b);
  // head
  k_head<<<1, 256, 0, stream>>>(h_f, h_b, W3, b3, W4, b4, out);
}
